// FCLSTM_56332791054969
// MI455X (gfx1250) — compile-verified, matrix operand order checked by reading, not run on hardware
//
#include <hip/hip_runtime.h>
#include <hip/hip_bf16.h>

// ---------------------------------------------------------------------------
// Fused cross-modal 2-layer LSTM for MI455X (gfx1250, wave32, WMMA bf16).
//
//  - Weights converted once to bf16 (~133 MB -> resident in 192 MB L2; the
//    128 recurrent steps stream weights from L2, not HBM).
//  - One concatenated bf16 weight [4H x Ktot] per modality/layer so each cell
//    pre-activation is a single multi-segment WMMA GEMM:
//      layer0: z = [xpad(s) | h1_crossA | h1_crossB | h0_prev] @ Wcat0^T
//      layer1: z = [h0 | h1_prev] @ Wcat1^T
//  - GEMM: workgroup tile 128x128, K-chunk 32, A/B tiles double-buffered in
//    LDS (one barrier per chunk; global prefetch of chunk k+1 issued before
//    the barrier overlaps chunk k's WMMAs). Segment loop is unrolled with
//    compile-time indices so descriptors stay in SGPRs (no scratch spills).
//  - Gate kernel: fused sigmoid/tanh, c kept fp32, h written bf16.
// ---------------------------------------------------------------------------

typedef __attribute__((ext_vector_type(16))) __bf16 v16bf;
typedef __attribute__((ext_vector_type(8)))  float  v8f;

#define Bq 256
#define Sq 128
#define Hq 1024
#define N4H 4096
#define LPITCH 48        // LDS row pitch in bf16 elems (96 B: 32B-aligned)
#define LBUF (128 * LPITCH)

struct Seg { const __bf16* A; int stride; int klen; };
struct ModeGemm { Seg seg[4]; const __bf16* W; float* Z; int Ktot; int nseg; };
struct GemmArgs { ModeGemm m[3]; };

struct GateArgs {
    const float* Z[3];
    float*       C[3];
    __bf16*      Hout[3];
    const float* bih[3];
    const float* bhh[3];
};

// ---------------------------- WMMA GEMM kernel -----------------------------
// Z[b, n] = sum over segments of A_seg[b, :] . W[n, segK...]   (fp32 accum)
__global__ __launch_bounds__(256) void fclstm_gemm(GemmArgs ga) {
    __shared__ __bf16 ldsA[2 * LBUF];
    __shared__ __bf16 ldsB[2 * LBUF];

    const ModeGemm mg = ga.m[blockIdx.z];

    const int tid   = threadIdx.x;
    const int lane  = tid & 31;
    const int wave  = tid >> 5;
    const int wm    = wave & 3;          // 4 waves along M
    const int wn    = wave >> 2;         // 2 waves along N
    const int lrow  = lane & 15;
    const int khalf = (lane >> 4) << 4;  // lanes 16..31 take K+16..K+31
    const int mblock = blockIdx.x * 128;
    const int nblock = blockIdx.y * 128;

    // LDS staging coords: 256 threads move 128 rows x 32 k of A and of B.
    const int srow = tid >> 1;           // 0..127
    const int skh  = (tid & 1) << 4;     // k-offset 0 or 16

    v8f acc[2][4];
#pragma unroll
    for (int i = 0; i < 2; ++i)
#pragma unroll
        for (int j = 0; j < 4; ++j)
#pragma unroll
            for (int e = 0; e < 8; ++e) acc[i][j][e] = 0.0f;

    // ---- prologue: stage chunk 0 (segment 0) into regs ----
    v16bf aS = *(const v16bf*)(mg.seg[0].A +
        (size_t)(mblock + srow) * mg.seg[0].stride + skh);
    v16bf bS = *(const v16bf*)(mg.W +
        (size_t)(nblock + srow) * mg.Ktot + skh);

    int kacc = 0;
    int p = 0;
#pragma unroll
    for (int s = 0; s < 4; ++s) {
        if (s < mg.nseg) {
            const Seg sg = mg.seg[s];            // compile-time index
            for (int kc = 0; kc < sg.klen; kc += 32) {
                __bf16* bufA = ldsA + p * LBUF;
                __bf16* bufB = ldsB + p * LBUF;
                *(v16bf*)(bufA + srow * LPITCH + skh) = aS;
                *(v16bf*)(bufB + srow * LPITCH + skh) = bS;

                // ---- prefetch chunk c+1 from global (L2) before barrier ----
                if (kc + 32 < sg.klen) {         // uniform scalar branch
                    aS = *(const v16bf*)(sg.A +
                        (size_t)(mblock + srow) * sg.stride + kc + 32 + skh);
                    bS = *(const v16bf*)(mg.W +
                        (size_t)(nblock + srow) * mg.Ktot + kacc + kc + 32 + skh);
                } else if (s + 1 < mg.nseg) {    // uniform scalar branch
                    const Seg nx = mg.seg[(s + 1) & 3];  // compile-time index
                    aS = *(const v16bf*)(nx.A +
                        (size_t)(mblock + srow) * nx.stride + skh);
                    bS = *(const v16bf*)(mg.W +
                        (size_t)(nblock + srow) * mg.Ktot + kacc + sg.klen + skh);
                }

                __syncthreads();   // chunk c visible; fences readers of c-1

                // ---- fragments from LDS ----
                v16bf afrag[2], bfrag[4];
#pragma unroll
                for (int i = 0; i < 2; ++i)
                    afrag[i] = *(const v16bf*)(bufA +
                        (wm * 32 + i * 16 + lrow) * LPITCH + khalf);
#pragma unroll
                for (int j = 0; j < 4; ++j)
                    bfrag[j] = *(const v16bf*)(bufB +
                        (wn * 64 + j * 16 + lrow) * LPITCH + khalf);

#pragma unroll
                for (int i = 0; i < 2; ++i)
#pragma unroll
                    for (int j = 0; j < 4; ++j)
                        acc[i][j] = __builtin_amdgcn_wmma_f32_16x16x32_bf16(
                            false, afrag[i], false, bfrag[j],
                            (short)0, acc[i][j], false, false);

                p ^= 1;
            }
            kacc += sg.klen;
        }
    }

    const int rowadd = (lane >> 4) * 8;  // C/D layout: lanes 16-31 hold M=8..15
#pragma unroll
    for (int i = 0; i < 2; ++i)
#pragma unroll
        for (int j = 0; j < 4; ++j)
#pragma unroll
            for (int r = 0; r < 8; ++r) {
                const int row = mblock + wm * 32 + i * 16 + rowadd + r;
                const int col = nblock + wn * 64 + j * 16 + lrow;
                mg.Z[(size_t)row * N4H + col] = acc[i][j][r];
            }
}

// ----------------------------- LSTM gate kernel ----------------------------
__global__ __launch_bounds__(256) void fclstm_gates(GateArgs ga) {
    const int mod = blockIdx.z;
    const int idx = blockIdx.x * 256 + threadIdx.x;  // 0 .. B*H-1
    const int b = idx >> 10;
    const int j = idx & (Hq - 1);
    const float* Z   = ga.Z[mod];
    const float* bih = ga.bih[mod];
    const float* bhh = ga.bhh[mod];
    const size_t zb = (size_t)b * N4H;
    const float zi = Z[zb + j]            + bih[j]          + bhh[j];
    const float zf = Z[zb + Hq + j]       + bih[Hq + j]     + bhh[Hq + j];
    const float zg = Z[zb + 2 * Hq + j]   + bih[2 * Hq + j] + bhh[2 * Hq + j];
    const float zo = Z[zb + 3 * Hq + j]   + bih[3 * Hq + j] + bhh[3 * Hq + j];
    const float ig = 1.0f / (1.0f + __expf(-zi));
    const float fg = 1.0f / (1.0f + __expf(-zf));
    const float og = 1.0f / (1.0f + __expf(-zo));
    const float cp = ga.C[mod][idx];
    const float cn = fg * cp + ig * tanhf(zg);
    ga.C[mod][idx] = cn;
    ga.Hout[mod][idx] = (__bf16)(og * tanhf(cn));
}

// -------------------------- one-time setup kernels -------------------------
__global__ void fclstm_zero_u32(unsigned int* p, long n) {
    const long i = (long)blockIdx.x * 256 + threadIdx.x;
    if (i < n) p[i] = 0u;
}

// x: [B, S, f] fp32  ->  out: [(s*B + b), f_pad] bf16, zero padded
__global__ void fclstm_conv_x(const float* __restrict__ x, __bf16* __restrict__ out,
                              int f, int fpad, int total) {
    const int idx = blockIdx.x * 256 + threadIdx.x;
    if (idx >= total) return;
    const int k = idx % fpad;
    const int r = idx / fpad;     // r = s*B + b
    const int s = r >> 8;         // /B (B==256)
    const int b = r & 255;
    float v = 0.0f;
    if (k < f) v = x[((size_t)b * Sq + s) * f + k];
    out[idx] = (__bf16)v;
}

// Wih0: [4H, f+2H], Whh0: [4H, H]  -> cat [4H, fpad + 3H] bf16
__global__ void fclstm_conv_w0(const float* __restrict__ Wih,
                               const float* __restrict__ Whh,
                               __bf16* __restrict__ out,
                               int f, int fpad, int ktot, int total) {
    const int idx = blockIdx.x * 256 + threadIdx.x;
    if (idx >= total) return;
    const int k = idx % ktot;
    const int n = idx / ktot;
    const int incols = f + 2 * Hq;
    float v;
    if (k < fpad)                v = (k < f) ? Wih[(size_t)n * incols + k] : 0.0f;
    else if (k < fpad + 2 * Hq)  v = Wih[(size_t)n * incols + f + (k - fpad)];
    else                         v = Whh[(size_t)n * Hq + (k - fpad - 2 * Hq)];
    out[idx] = (__bf16)v;
}

// Wih1: [4H, H], Whh1: [4H, H] -> cat [4H, 2H] bf16 (total = 4096*2048)
__global__ void fclstm_conv_w1(const float* __restrict__ Wih,
                               const float* __restrict__ Whh,
                               __bf16* __restrict__ out) {
    const int idx = blockIdx.x * 256 + threadIdx.x;
    const int k = idx & 2047;
    const int n = idx >> 11;
    const float v = (k < Hq) ? Wih[(size_t)n * Hq + k]
                             : Whh[(size_t)n * Hq + (k - Hq)];
    out[idx] = (__bf16)v;
}

// ------------------------------- final dense -------------------------------
__global__ void fclstm_dense(const __bf16* __restrict__ ht,
                             const __bf16* __restrict__ hv,
                             const __bf16* __restrict__ ha,
                             const float* __restrict__ Wd,
                             const float* __restrict__ bd,
                             float* __restrict__ out) {
    const int b = threadIdx.x;  // 256 threads
    float s = bd[0];
    for (int j = 0; j < Hq; ++j) {
        s += (float)ht[(size_t)b * Hq + j] * Wd[j];
        s += (float)hv[(size_t)b * Hq + j] * Wd[Hq + j];
        s += (float)ha[(size_t)b * Hq + j] * Wd[2 * Hq + j];
    }
    out[b] = s;
}

// ------------------------------- host driver -------------------------------
extern "C" void kernel_launch(void* const* d_in, const int* in_sizes, int n_in,
                              void* d_out, int out_size, void* d_ws, size_t ws_size,
                              hipStream_t stream) {
    (void)in_sizes; (void)n_in; (void)out_size; (void)ws_size;

    const int FEAT[3] = {300, 35, 74};
    const int FPAD[3] = {320, 64, 96};
    const int KT0[3]  = {320 + 3 * Hq, 64 + 3 * Hq, 96 + 3 * Hq};
    const int cross[3][2] = {{1, 2}, {0, 2}, {0, 1}};  // t:(v,a) v:(t,a) a:(t,v)

    char* ws = (char*)d_ws;
    size_t off = 0;
    auto take = [&](size_t bytes) -> char* {
        char* p = ws + off;
        off = (off + bytes + 255) & ~(size_t)255;
        return p;
    };

    // State buffers (zeroed every call), contiguous region.
    char* stateBase = ws + off;
    __bf16* h0[3]; __bf16* h1[3]; float* c0[3]; float* c1[3];
    for (int m = 0; m < 3; ++m) {
        h0[m] = (__bf16*)take((size_t)Bq * Hq * 2);
        h1[m] = (__bf16*)take((size_t)Bq * Hq * 2);
    }
    for (int m = 0; m < 3; ++m) {
        c0[m] = (float*)take((size_t)Bq * Hq * 4);
        c1[m] = (float*)take((size_t)Bq * Hq * 4);
    }
    const size_t stateBytes = (size_t)((ws + off) - stateBase);

    float* z[3];
    for (int m = 0; m < 3; ++m) z[m] = (float*)take((size_t)Bq * N4H * 4);
    __bf16* xpad[3];
    for (int m = 0; m < 3; ++m) xpad[m] = (__bf16*)take((size_t)Sq * Bq * FPAD[m] * 2);
    __bf16* W0[3];
    for (int m = 0; m < 3; ++m) W0[m] = (__bf16*)take((size_t)N4H * KT0[m] * 2);
    __bf16* W1[3];
    for (int m = 0; m < 3; ++m) W1[m] = (__bf16*)take((size_t)N4H * 2 * Hq * 2);

    // Input pointer map (setup_inputs order).
    const float* xin[3] = {(const float*)d_in[0], (const float*)d_in[1], (const float*)d_in[2]};
    const float *Wih0[3], *Whh0[3], *bih0[3], *bhh0[3];
    const float *Wih1[3], *Whh1[3], *bih1[3], *bhh1[3];
    for (int m = 0; m < 3; ++m) {
        const int base = 3 + 8 * m;
        Wih0[m] = (const float*)d_in[base + 0];
        Whh0[m] = (const float*)d_in[base + 1];
        bih0[m] = (const float*)d_in[base + 2];
        bhh0[m] = (const float*)d_in[base + 3];
        Wih1[m] = (const float*)d_in[base + 4];
        Whh1[m] = (const float*)d_in[base + 5];
        bih1[m] = (const float*)d_in[base + 6];
        bhh1[m] = (const float*)d_in[base + 7];
    }
    const float* Wd = (const float*)d_in[27];
    const float* bd = (const float*)d_in[28];
    float* out = (float*)d_out;

    // ---- setup: zero states, convert x & weights to bf16 layouts ----
    {
        long n = (long)(stateBytes / 4);
        fclstm_zero_u32<<<dim3((unsigned)((n + 255) / 256)), dim3(256), 0, stream>>>(
            (unsigned int*)stateBase, n);
    }
    for (int m = 0; m < 3; ++m) {
        const int total = Sq * Bq * FPAD[m];
        fclstm_conv_x<<<dim3((total + 255) / 256), dim3(256), 0, stream>>>(
            xin[m], xpad[m], FEAT[m], FPAD[m], total);
    }
    for (int m = 0; m < 3; ++m) {
        const int total = N4H * KT0[m];
        fclstm_conv_w0<<<dim3((total + 255) / 256), dim3(256), 0, stream>>>(
            Wih0[m], Whh0[m], W0[m], FEAT[m], FPAD[m], KT0[m], total);
    }
    for (int m = 0; m < 3; ++m) {
        const int total = N4H * 2 * Hq;  // 8388608, exact multiple of 256
        fclstm_conv_w1<<<dim3(total / 256), dim3(256), 0, stream>>>(
            Wih1[m], Whh1[m], W1[m]);
    }

    // ---- sequential scan over S steps ----
    const dim3 gemmGrid(Bq / 128, N4H / 128, 3);   // (2, 32, 3)
    const dim3 gateGrid(Bq * Hq / 256, 1, 3);      // (1024, 1, 3)

    for (int s = 0; s < Sq; ++s) {
        // Layer 0: z_m = [xpad_m(s) | h1_crossA | h1_crossB | h0_m] @ W0_m^T
        GemmArgs g0;
        for (int m = 0; m < 3; ++m) {
            g0.m[m].seg[0] = Seg{xpad[m] + (size_t)s * Bq * FPAD[m], FPAD[m], FPAD[m]};
            g0.m[m].seg[1] = Seg{h1[cross[m][0]], Hq, Hq};
            g0.m[m].seg[2] = Seg{h1[cross[m][1]], Hq, Hq};
            g0.m[m].seg[3] = Seg{h0[m], Hq, Hq};
            g0.m[m].W = W0[m]; g0.m[m].Z = z[m]; g0.m[m].Ktot = KT0[m]; g0.m[m].nseg = 4;
        }
        fclstm_gemm<<<gemmGrid, dim3(256), 0, stream>>>(g0);

        GateArgs ga0;
        for (int m = 0; m < 3; ++m) {
            ga0.Z[m] = z[m]; ga0.C[m] = c0[m]; ga0.Hout[m] = h0[m];
            ga0.bih[m] = bih0[m]; ga0.bhh[m] = bhh0[m];
        }
        fclstm_gates<<<gateGrid, dim3(256), 0, stream>>>(ga0);

        // Layer 1: z_m = [h0_m | h1_m_prev] @ W1_m^T
        GemmArgs g1;
        for (int m = 0; m < 3; ++m) {
            g1.m[m].seg[0] = Seg{h0[m], Hq, Hq};
            g1.m[m].seg[1] = Seg{h1[m], Hq, Hq};
            g1.m[m].seg[2] = Seg{h0[m], Hq, 0};
            g1.m[m].seg[3] = Seg{h0[m], Hq, 0};
            g1.m[m].W = W1[m]; g1.m[m].Z = z[m]; g1.m[m].Ktot = 2 * Hq; g1.m[m].nseg = 2;
        }
        fclstm_gemm<<<gemmGrid, dim3(256), 0, stream>>>(g1);

        GateArgs ga1;
        for (int m = 0; m < 3; ++m) {
            ga1.Z[m] = z[m]; ga1.C[m] = c1[m]; ga1.Hout[m] = h1[m];
            ga1.bih[m] = bih1[m]; ga1.bhh[m] = bhh1[m];
        }
        fclstm_gates<<<gateGrid, dim3(256), 0, stream>>>(ga1);
    }

    // ---- final dense head ----
    fclstm_dense<<<dim3(1), dim3(256), 0, stream>>>(h1[0], h1[1], h1[2], Wd, bd, out);
}
